// Baseline_66056597012621
// MI455X (gfx1250) — compile-verified, matrix operand order checked by reading, not run on hardware
//
#include <hip/hip_runtime.h>
#include <hip/hip_bf16.h>
#include <math.h>

// ---------------------------------------------------------------------------
// Problem constants (from reference)
// ---------------------------------------------------------------------------
#define BB      2
#define LL      512
#define DD      768
#define DDI     1536
#define TWO_DI  3072
#define NN      16
#define RR      48
#define RPAD    64
#define KK      4
#define VV      50280
#define NLAY    4
#define MTOK    2048          // 2 runs * B * L
#define NPROJ   80            // R + 2N (logical)
#define NPROJP  128           // padded to multiple of 64 for branch-free GEMM
#define NCOL    256           // vocab cols per logits block
#define NCHUNK  ((VV + NCOL - 1) / NCOL)   // 197

typedef __attribute__((ext_vector_type(16))) _Float16 v16h;
typedef __attribute__((ext_vector_type(8)))  float    v8f;

__device__ __forceinline__ float silu_f(float x) { return x / (1.0f + __expf(-x)); }
__device__ __forceinline__ float softplus_f(float x) {
    return (x > 20.0f) ? x : log1pf(__expf(x));
}

// Load one 16-half WMMA fragment (A row / Bt row) at k-offset `k`.
// Layout per CDNA5 ISA: elems 0..7 -> K = 8*half + j, elems 8..15 -> K = 16 + 8*half + j.
__device__ __forceinline__ v16h load_frag(const _Float16* __restrict__ p, int k) {
    v16h f;
#pragma unroll
    for (int j = 0; j < 8; ++j) {
        f[j]     = p[k + j];
        f[8 + j] = p[k + 16 + j];
    }
    return f;
}

// ---------------------------------------------------------------------------
// f32 -> f16 elementwise convert (embed table)
// ---------------------------------------------------------------------------
__global__ void cvt_f16_kernel(const float* __restrict__ src, _Float16* __restrict__ dst,
                               size_t n) {
    size_t i = (size_t)blockIdx.x * blockDim.x + threadIdx.x;
    if (i < n) dst[i] = (_Float16)src[i];
}

// W[K,N] f32  ->  Wt[Npad,Kpad] f16 (zero padded in both K and N)
__global__ void xpose_f16_kernel(const float* __restrict__ src, _Float16* __restrict__ dst,
                                 int K, int N, int Kpad, int Npad) {
    size_t i = (size_t)blockIdx.x * blockDim.x + threadIdx.x;
    size_t tot = (size_t)Npad * Kpad;
    if (i >= tot) return;
    int k = (int)(i % Kpad);
    int n = (int)(i / Kpad);
    dst[i] = (k < K && n < N) ? (_Float16)src[(size_t)k * N + n] : (_Float16)0.0f;
}

// ---------------------------------------------------------------------------
// Embedding gather: x[t,:] = embed[ids] * mask    (4 chains of 512 tokens)
// ---------------------------------------------------------------------------
__global__ void gather_embed_kernel(const float* __restrict__ embed,
                                    const int* __restrict__ qids, const int* __restrict__ qmask,
                                    const int* __restrict__ aids, const int* __restrict__ amask,
                                    float* __restrict__ x) {
    int t = blockIdx.x;               // 0..2047
    int chain = t >> 9, pos = t & 511;
    const int* ids; const int* msk; int b;
    if (chain < 2) { ids = qids; msk = qmask; b = chain; }
    else           { ids = aids; msk = amask; b = chain - 2; }
    int id = ids[b * LL + pos];
    float m = (float)msk[b * LL + pos];
    const float* erow = embed + (size_t)id * DD;
    float* xrow = x + (size_t)t * DD;
    for (int d = threadIdx.x; d < DD; d += blockDim.x) xrow[d] = erow[d] * m;
}

// ---------------------------------------------------------------------------
// Labels / loss weights per global token row
// ---------------------------------------------------------------------------
__global__ void labels_kernel(const int* __restrict__ qids, const int* __restrict__ qmask,
                              const int* __restrict__ aids, const int* __restrict__ amask,
                              int* __restrict__ labels, float* __restrict__ wts) {
    int t = blockIdx.x * blockDim.x + threadIdx.x;
    if (t >= MTOK) return;
    int chain = t >> 9, pos = t & 511;
    if (pos >= LL - 1) { labels[t] = -1; wts[t] = 0.0f; return; }
    const int* ids; const int* msk; int b;
    if (chain < 2) { ids = qids; msk = qmask; b = chain; }
    else           { ids = aids; msk = amask; b = chain - 2; }
    labels[t] = ids[b * LL + pos + 1];
    wts[t]    = (float)msk[b * LL + pos + 1];
}

// ---------------------------------------------------------------------------
// RMSNorm -> f16 output (A operand for the following GEMM)
// ---------------------------------------------------------------------------
__global__ void rmsnorm_f16_kernel(const float* __restrict__ x, const float* __restrict__ w,
                                   _Float16* __restrict__ out) {
    int t = blockIdx.x;
    __shared__ float red[256];
    const float* xr = x + (size_t)t * DD;
    float s = 0.0f;
    for (int d = threadIdx.x; d < DD; d += blockDim.x) { float v = xr[d]; s += v * v; }
    red[threadIdx.x] = s;
    __syncthreads();
    for (int o = 128; o > 0; o >>= 1) {
        if ((int)threadIdx.x < o) red[threadIdx.x] += red[threadIdx.x + o];
        __syncthreads();
    }
    float rs = rsqrtf(red[0] / (float)DD + 1e-5f);
    _Float16* orow = out + (size_t)t * DD;
    for (int d = threadIdx.x; d < DD; d += blockDim.x)
        orow[d] = (_Float16)(xr[d] * rs * w[d]);
}

// ---------------------------------------------------------------------------
// Register-blocked WMMA GEMM: each wave computes a 32x64 macro-tile
// (2 M-frags x 4 N-frags -> 8 WMMAs per 32-wide k-step). Branch-free hot loop.
// C[M,N] = epilogue( A[M,K]f16 @ Wt[N,K]f16 + bias ).
// Requires K % 32 == 0, M % 32 == 0, N % 64 == 0.
// act: 0=none, 1=silu, 2=softplus.   Cin != null -> residual add.
// ---------------------------------------------------------------------------
__global__ void __launch_bounds__(32)
gemm_wmma_kernel(const _Float16* __restrict__ A, const _Float16* __restrict__ Bt,
                 const float* __restrict__ bias, const float* __restrict__ Cin,
                 float* __restrict__ Cout, _Float16* __restrict__ Out16,
                 int M, int N, int K, int act) {
    int n0 = blockIdx.x * 64;
    int m0 = blockIdx.y * 32;
    int lane = threadIdx.x & 31;
    int hlf  = lane >> 4;
    int idx  = lane & 15;

    const _Float16* ap0 = A + (size_t)(m0 + idx) * K + hlf * 8;
    const _Float16* ap1 = A + (size_t)(m0 + 16 + idx) * K + hlf * 8;
    const _Float16* bp0 = Bt + (size_t)(n0 + idx) * K + hlf * 8;
    const _Float16* bp1 = Bt + (size_t)(n0 + 16 + idx) * K + hlf * 8;
    const _Float16* bp2 = Bt + (size_t)(n0 + 32 + idx) * K + hlf * 8;
    const _Float16* bp3 = Bt + (size_t)(n0 + 48 + idx) * K + hlf * 8;

    v8f c[2][4] = {};
    for (int k = 0; k < K; k += 32) {
        v16h a0 = load_frag(ap0, k);
        v16h a1 = load_frag(ap1, k);
        v16h b0 = load_frag(bp0, k);
        v16h b1 = load_frag(bp1, k);
        v16h b2 = load_frag(bp2, k);
        v16h b3 = load_frag(bp3, k);
        c[0][0] = __builtin_amdgcn_wmma_f32_16x16x32_f16(false, a0, false, b0, (short)0, c[0][0], false, false);
        c[1][0] = __builtin_amdgcn_wmma_f32_16x16x32_f16(false, a1, false, b0, (short)0, c[1][0], false, false);
        c[0][1] = __builtin_amdgcn_wmma_f32_16x16x32_f16(false, a0, false, b1, (short)0, c[0][1], false, false);
        c[1][1] = __builtin_amdgcn_wmma_f32_16x16x32_f16(false, a1, false, b1, (short)0, c[1][1], false, false);
        c[0][2] = __builtin_amdgcn_wmma_f32_16x16x32_f16(false, a0, false, b2, (short)0, c[0][2], false, false);
        c[1][2] = __builtin_amdgcn_wmma_f32_16x16x32_f16(false, a1, false, b2, (short)0, c[1][2], false, false);
        c[0][3] = __builtin_amdgcn_wmma_f32_16x16x32_f16(false, a0, false, b3, (short)0, c[0][3], false, false);
        c[1][3] = __builtin_amdgcn_wmma_f32_16x16x32_f16(false, a1, false, b3, (short)0, c[1][3], false, false);
    }
#pragma unroll
    for (int mi = 0; mi < 2; ++mi) {
#pragma unroll
        for (int j = 0; j < 4; ++j) {
#pragma unroll
            for (int r = 0; r < 8; ++r) {
                int gm = m0 + mi * 16 + r + 8 * hlf;
                int gn = n0 + j * 16 + idx;
                float v = c[mi][j][r];
                if (bias) v += bias[gn];
                if (act == 1)      v = silu_f(v);
                else if (act == 2) v = softplus_f(v);
                size_t o = (size_t)gm * N + gn;
                if (Cin) v += Cin[o];
                Cout[o] = v;
                if (Out16) Out16[o] = (_Float16)v;
            }
        }
    }
}

// ---------------------------------------------------------------------------
// Causal depthwise conv (K=4) + bias + SiLU, per 512-token chain
// ---------------------------------------------------------------------------
__global__ void conv_silu_kernel(const float* __restrict__ xz, const float* __restrict__ cw,
                                 const float* __restrict__ cb,
                                 float* __restrict__ uc, _Float16* __restrict__ uc16) {
    size_t g = (size_t)blockIdx.x * blockDim.x + threadIdx.x;
    if (g >= (size_t)MTOK * DDI) return;
    int di = (int)(g % DDI);
    int t  = (int)(g / DDI);
    int pos = t & 511;
    float acc = cb[di];
#pragma unroll
    for (int k = 0; k < KK; ++k) {
        int off = k - (KK - 1);
        if (pos + off >= 0)
            acc += xz[(size_t)(t + off) * TWO_DI + di] * cw[di * KK + k];
    }
    float s = silu_f(acc);
    uc[g]   = s;
    uc16[g] = (_Float16)s;
}

// dtA16[t, r<64] = (r<48) ? proj[t,r] : 0     (proj has stride NPROJP)
__global__ void prep_dtA_kernel(const float* __restrict__ proj, _Float16* __restrict__ dtA) {
    size_t g = (size_t)blockIdx.x * blockDim.x + threadIdx.x;
    if (g >= (size_t)MTOK * RPAD) return;
    int r = (int)(g & (RPAD - 1));
    int t = (int)(g >> 6);
    dtA[g] = (r < RR) ? (_Float16)proj[(size_t)t * NPROJP + r] : (_Float16)0.0f;
}

// ---------------------------------------------------------------------------
// Selective scan: h[n] = exp(-dt*exp(Alog[n]))*h[n] + dt*u*B[n];  y = h.C
// One thread per (chain, channel). Fuses +u*D and silu(z) gate; emits f16 y.
// ---------------------------------------------------------------------------
__global__ void scan_kernel(const float* __restrict__ dt, const float* __restrict__ uc,
                            const float* __restrict__ proj, const float* __restrict__ xz,
                            const float* __restrict__ Alog, const float* __restrict__ Dp,
                            _Float16* __restrict__ y16) {
    int g = blockIdx.x * blockDim.x + threadIdx.x;      // 0 .. 4*1536-1
    if (g >= 4 * DDI) return;
    int di = g % DDI;
    int chain = g / DDI;

    float eA[NN];
#pragma unroll
    for (int n = 0; n < NN; ++n) eA[n] = __expf(Alog[di * NN + n]);
    float Dv = Dp[di];

    float h[NN];
#pragma unroll
    for (int n = 0; n < NN; ++n) h[n] = 0.0f;

    for (int pos = 0; pos < LL; ++pos) {
        int t = chain * LL + pos;
        float d = dt[(size_t)t * DDI + di];
        float u = uc[(size_t)t * DDI + di];
        float du = d * u;
        const float* pr = proj + (size_t)t * NPROJP;
        float y = 0.0f;
#pragma unroll
        for (int n = 0; n < NN; ++n) {
            float a = __expf(-d * eA[n]);
            h[n] = a * h[n] + du * pr[RR + n];
            y += h[n] * pr[RR + NN + n];
        }
        y += u * Dv;
        float z = xz[(size_t)t * TWO_DI + DDI + di];
        y *= silu_f(z);
        y16[(size_t)t * DDI + di] = (_Float16)y;
    }
}

// ---------------------------------------------------------------------------
// Logits tile + streaming-softmax partials.
// Block = 4 waves; 32 token rows x 256 vocab cols; A tile staged in LDS.
// Each wave: 2 M-frags x 4 N-subtiles; B fragment reused by 2 WMMAs.
// ---------------------------------------------------------------------------
__global__ void __launch_bounds__(128)
logits_kernel(const _Float16* __restrict__ xf16, const _Float16* __restrict__ emb16,
              const int* __restrict__ labels, float* __restrict__ labelLogit,
              float* __restrict__ pMax, float* __restrict__ pSum) {
    __shared__ _Float16 As[32 * DD];     // 48 KB
    __shared__ float    Ls[32 * NCOL];   // 32 KB
    int m0 = blockIdx.y * 32;
    int chunk = blockIdx.x;
    int col0base = chunk * NCOL;

    for (int i = threadIdx.x; i < 32 * DD; i += 128)
        As[i] = xf16[(size_t)m0 * DD + i];
    __syncthreads();

    int wave = threadIdx.x >> 5;
    int lane = threadIdx.x & 31;
    int hlf  = lane >> 4;
    int idx  = lane & 15;

    const _Float16* ap0 = As + idx * DD + hlf * 8;
    const _Float16* ap1 = As + (16 + idx) * DD + hlf * 8;

    for (int s = 0; s < 4; ++s) {
        int nsub = wave * 4 + s;
        int col = col0base + nsub * 16 + idx;
        bool ok = (col < VV);
        const _Float16* bp = emb16 + (size_t)(ok ? col : 0) * DD + hlf * 8;
        v8f c0 = {}, c1 = {};
        for (int k = 0; k < DD; k += 32) {
            v16h b  = load_frag(bp, k);
            v16h a0 = load_frag(ap0, k);
            v16h a1 = load_frag(ap1, k);
            c0 = __builtin_amdgcn_wmma_f32_16x16x32_f16(false, a0, false, b,
                                                        (short)0, c0, false, false);
            c1 = __builtin_amdgcn_wmma_f32_16x16x32_f16(false, a1, false, b,
                                                        (short)0, c1, false, false);
        }
#pragma unroll
        for (int r = 0; r < 8; ++r) {
            int ml0 = r + 8 * hlf;
            int ml1 = 16 + ml0;
            float v0 = ok ? c0[r] : -3.0e38f;
            float v1 = ok ? c1[r] : -3.0e38f;
            Ls[ml0 * NCOL + nsub * 16 + idx] = v0;
            Ls[ml1 * NCOL + nsub * 16 + idx] = v1;
            if (ok && labels[m0 + ml0] == col) labelLogit[m0 + ml0] = c0[r];
            if (ok && labels[m0 + ml1] == col) labelLogit[m0 + ml1] = c1[r];
        }
    }
    __syncthreads();

    if (threadIdx.x < 32) {
        int row = threadIdx.x;
        const float* lr = Ls + row * NCOL;
        float mx = -3.0e38f;
        for (int j = 0; j < NCOL; ++j) mx = fmaxf(mx, lr[j]);
        float s = 0.0f;
        for (int j = 0; j < NCOL; ++j) {
            float v = lr[j];
            if (v > -1.0e38f) s += __expf(v - mx);
        }
        pMax[(size_t)(m0 + row) * NCHUNK + chunk] = mx;
        pSum[(size_t)(m0 + row) * NCHUNK + chunk] = s;
    }
}

// ---------------------------------------------------------------------------
// Combine streaming-softmax partials -> LSE -> NLL -> masked means -> loss
// ---------------------------------------------------------------------------
__global__ void loss_reduce_kernel(const float* __restrict__ pMax, const float* __restrict__ pSum,
                                   const float* __restrict__ labelLogit,
                                   const int* __restrict__ labels, const float* __restrict__ wts,
                                   float* __restrict__ out) {
    __shared__ float sq[256], sa[256], cq[256], ca[256];
    float aq = 0.0f, aa = 0.0f, wq = 0.0f, wa = 0.0f;
    for (int row = threadIdx.x; row < MTOK; row += 256) {
        if (labels[row] < 0) continue;
        const float* pm = pMax + (size_t)row * NCHUNK;
        const float* ps = pSum + (size_t)row * NCHUNK;
        float mx = -3.0e38f;
        for (int j = 0; j < NCHUNK; ++j) mx = fmaxf(mx, pm[j]);
        float s = 0.0f;
        for (int j = 0; j < NCHUNK; ++j) {
            float m = pm[j];
            if (m > -1.0e37f) s += ps[j] * __expf(m - mx);
        }
        float lse = mx + __logf(s);
        float nll = lse - labelLogit[row];
        float w = wts[row];
        if (row < 2 * LL) { aq += w * nll; wq += w; }   // question chains 0,1
        else              { aa += w * nll; wa += w; }   // answer chains 2,3
    }
    sq[threadIdx.x] = aq; sa[threadIdx.x] = aa;
    cq[threadIdx.x] = wq; ca[threadIdx.x] = wa;
    __syncthreads();
    for (int o = 128; o > 0; o >>= 1) {
        if ((int)threadIdx.x < o) {
            sq[threadIdx.x] += sq[threadIdx.x + o];
            sa[threadIdx.x] += sa[threadIdx.x + o];
            cq[threadIdx.x] += cq[threadIdx.x + o];
            ca[threadIdx.x] += ca[threadIdx.x + o];
        }
        __syncthreads();
    }
    if (threadIdx.x == 0)
        out[0] = sa[0] / fmaxf(ca[0], 1.0f) + sq[0] / fmaxf(cq[0], 1.0f);
}

// ---------------------------------------------------------------------------
// Host-side launcher
// ---------------------------------------------------------------------------
extern "C" void kernel_launch(void* const* d_in, const int* in_sizes, int n_in,
                              void* d_out, int out_size, void* d_ws, size_t ws_size,
                              hipStream_t stream) {
    (void)in_sizes; (void)n_in; (void)out_size; (void)ws_size;

    const int*   q_ids   = (const int*)d_in[0];
    const int*   q_mask  = (const int*)d_in[1];
    const int*   a_ids   = (const int*)d_in[2];
    const int*   a_mask  = (const int*)d_in[3];
    const float* embed   = (const float*)d_in[4];
    const float* ipw     = (const float*)d_in[5];
    const float* convw   = (const float*)d_in[6];
    const float* convb   = (const float*)d_in[7];
    const float* xpw     = (const float*)d_in[8];
    const float* dtw     = (const float*)d_in[9];
    const float* dtb     = (const float*)d_in[10];
    const float* A_log   = (const float*)d_in[11];
    const float* D_param = (const float*)d_in[12];
    const float* opw     = (const float*)d_in[13];
    const float* norm_w  = (const float*)d_in[14];
    const float* norm_fw = (const float*)d_in[15];
    float* out = (float*)d_out;

    // ---- workspace carve-up (deterministic) ----
    size_t off = 0;
    char* base = (char*)d_ws;
    auto alloc = [&](size_t bytes) -> char* {
        char* p = base + off;
        off += (bytes + 255) & ~(size_t)255;
        return p;
    };
    _Float16* emb16 = (_Float16*)alloc((size_t)VV * DD * 2);
    _Float16* ipwT[NLAY]; _Float16* xpwT[NLAY]; _Float16* dtwT[NLAY]; _Float16* opwT[NLAY];
    for (int l = 0; l < NLAY; ++l) {
        ipwT[l] = (_Float16*)alloc((size_t)TWO_DI * DD * 2);   // [3072,768]
        xpwT[l] = (_Float16*)alloc((size_t)NPROJP * DDI * 2);  // [128,1536] (rows 80.. zero)
        dtwT[l] = (_Float16*)alloc((size_t)DDI * RPAD * 2);    // [1536,64]
        opwT[l] = (_Float16*)alloc((size_t)DD * DDI * 2);      // [768,1536]
    }
    float*    xbuf   = (float*)   alloc((size_t)MTOK * DD * 4);
    _Float16* h16    = (_Float16*)alloc((size_t)MTOK * DD * 2);
    float*    xz     = (float*)   alloc((size_t)MTOK * TWO_DI * 4);
    float*    ucb    = (float*)   alloc((size_t)MTOK * DDI * 4);
    _Float16* uc16   = (_Float16*)alloc((size_t)MTOK * DDI * 2);
    float*    projb  = (float*)   alloc((size_t)MTOK * NPROJP * 4);
    _Float16* dtA16  = (_Float16*)alloc((size_t)MTOK * RPAD * 2);
    float*    dtbuf  = (float*)   alloc((size_t)MTOK * DDI * 4);
    _Float16* y16    = (_Float16*)alloc((size_t)MTOK * DDI * 2);
    _Float16* xf16   = (_Float16*)alloc((size_t)MTOK * DD * 2);
    float*    pMax   = (float*)   alloc((size_t)MTOK * NCHUNK * 4);
    float*    pSum   = (float*)   alloc((size_t)MTOK * NCHUNK * 4);
    int*      labels = (int*)     alloc((size_t)MTOK * 4);
    float*    wts    = (float*)   alloc((size_t)MTOK * 4);
    float*    labLog = (float*)   alloc((size_t)MTOK * 4);

    auto cdiv = [](size_t a, size_t b) { return (unsigned)((a + b - 1) / b); };

    // ---- prep: f16 weights ----
    {
        size_t n = (size_t)VV * DD;
        cvt_f16_kernel<<<cdiv(n, 256), 256, 0, stream>>>(embed, emb16, n);
    }
    for (int l = 0; l < NLAY; ++l) {
        xpose_f16_kernel<<<cdiv((size_t)TWO_DI * DD, 256), 256, 0, stream>>>(
            ipw + (size_t)l * DD * TWO_DI, ipwT[l], DD, TWO_DI, DD, TWO_DI);
        xpose_f16_kernel<<<cdiv((size_t)NPROJP * DDI, 256), 256, 0, stream>>>(
            xpw + (size_t)l * DDI * NPROJ, xpwT[l], DDI, NPROJ, DDI, NPROJP);
        xpose_f16_kernel<<<cdiv((size_t)DDI * RPAD, 256), 256, 0, stream>>>(
            dtw + (size_t)l * RR * DDI, dtwT[l], RR, DDI, RPAD, DDI);
        xpose_f16_kernel<<<cdiv((size_t)DD * DDI, 256), 256, 0, stream>>>(
            opw + (size_t)l * DDI * DD, opwT[l], DDI, DD, DDI, DD);
    }
    labels_kernel<<<cdiv(MTOK, 256), 256, 0, stream>>>(q_ids, q_mask, a_ids, a_mask,
                                                       labels, wts);
    gather_embed_kernel<<<MTOK, 256, 0, stream>>>(embed, q_ids, q_mask, a_ids, a_mask, xbuf);

    // ---- 4 Mamba blocks ----
    for (int l = 0; l < NLAY; ++l) {
        rmsnorm_f16_kernel<<<MTOK, 256, 0, stream>>>(xbuf, norm_w + (size_t)l * DD, h16);

        dim3 gIn(TWO_DI / 64, MTOK / 32);
        gemm_wmma_kernel<<<gIn, 32, 0, stream>>>(h16, ipwT[l], nullptr, nullptr,
                                                 xz, nullptr, MTOK, TWO_DI, DD, 0);

        conv_silu_kernel<<<cdiv((size_t)MTOK * DDI, 256), 256, 0, stream>>>(
            xz, convw + (size_t)l * DDI * KK, convb + (size_t)l * DDI, ucb, uc16);

        dim3 gXp(NPROJP / 64, MTOK / 32);
        gemm_wmma_kernel<<<gXp, 32, 0, stream>>>(uc16, xpwT[l], nullptr, nullptr,
                                                 projb, nullptr, MTOK, NPROJP, DDI, 0);

        prep_dtA_kernel<<<cdiv((size_t)MTOK * RPAD, 256), 256, 0, stream>>>(projb, dtA16);

        dim3 gDt(DDI / 64, MTOK / 32);
        gemm_wmma_kernel<<<gDt, 32, 0, stream>>>(dtA16, dtwT[l], dtb + (size_t)l * DDI,
                                                 nullptr, dtbuf, nullptr,
                                                 MTOK, DDI, RPAD, 2 /*softplus*/);

        scan_kernel<<<cdiv(4 * DDI, 256), 256, 0, stream>>>(
            dtbuf, ucb, projb, xz,
            A_log + (size_t)l * DDI * NN, D_param + (size_t)l * DDI, y16);

        dim3 gOut(DD / 64, MTOK / 32);
        gemm_wmma_kernel<<<gOut, 32, 0, stream>>>(y16, opwT[l], nullptr, xbuf /*residual*/,
                                                  xbuf, nullptr, MTOK, DD, DDI, 0);
    }

    // ---- final norm + vocab logits + loss ----
    rmsnorm_f16_kernel<<<MTOK, 256, 0, stream>>>(xbuf, norm_fw, xf16);

    dim3 gLg(NCHUNK, MTOK / 32);
    logits_kernel<<<gLg, 128, 0, stream>>>(xf16, emb16, labels, labLog, pMax, pSum);

    loss_reduce_kernel<<<1, 256, 0, stream>>>(pMax, pSum, labLog, labels, wts, out);
}
